// MultiHeadAttentionBlock_33792802685107
// MI455X (gfx1250) — matrix operand order repaired and checked by reading, not compiled, not run on hardware
//
#include <hip/hip_runtime.h>
#include <hip/hip_bf16.h>

// ---------------------------------------------------------------------------
// MHA block for MI455X (gfx1250, wave32).
//  - all matmuls: v_wmma_f32_16x16x32_bf16, fp32 accumulate
//  - GEMM tile staging: global_load_async_to_lds_b128 (ASYNCcnt), double-buffer
//  - attention K/V tile staging: Tensor Data Mover (tensor_load_to_lds,
//    TENSORcnt), double-buffer
//  - attention computes S^T = K*Q^T and O^T = V^T*P^T so softmax statistics
//    are per-lane scalars (wave32-friendly) and outputs pack into b128 stores
// ---------------------------------------------------------------------------

typedef __attribute__((ext_vector_type(16))) __bf16 v16bf;
typedef __attribute__((ext_vector_type(8)))  __bf16 v8bf;
typedef __attribute__((ext_vector_type(8)))  float  v8f;
typedef __attribute__((ext_vector_type(4)))  unsigned int u32x4;
typedef __attribute__((ext_vector_type(8)))  int          i32x8;
typedef __attribute__((ext_vector_type(4)))  int          i32x4;

constexpr int Bc  = 2;
constexpr int Sc  = 2048;
constexpr int Dc  = 1024;
constexpr int Hc  = 16;
constexpr int DKc = 64;
constexpr int Mtot = Bc * Sc;   // 4096 tokens

#define DEV __device__ __forceinline__

DEV unsigned short f2bf(float x) {
  union { float f; unsigned u; } c; c.f = x;
  unsigned r = c.u + 0x7FFFu + ((c.u >> 16) & 1u);   // round-to-nearest-even
  return (unsigned short)(r >> 16);
}

// Pack two floats into a bf16 pair.  Prefer the packed-convert VALU op.
DEV unsigned pack2bf(float a, float b) {
#if __has_builtin(__builtin_amdgcn_cvt_pk_bf16_f32)
  auto r = __builtin_amdgcn_cvt_pk_bf16_f32(a, b);
  union { decltype(r) v; unsigned u; } c; c.v = r;
  return c.u;
#else
  return (unsigned)f2bf(a) | ((unsigned)f2bf(b) << 16);
#endif
}

DEV v16bf join16(v8bf lo, v8bf hi) {
  return __builtin_shufflevector(lo, hi, 0,1,2,3,4,5,6,7,8,9,10,11,12,13,14,15);
}

// A-matrix 16x32 bf16 fragment: lane = row m (l%16), half g=l/16,
// contiguous K = g*16 .. g*16+15.
DEV v16bf loadA(const __bf16* row, int hg) {
  const __bf16* p = row + hg * 16;
  return join16(*(const v8bf*)p, *(const v8bf*)(p + 8));
}
// B-matrix 32x16 bf16 fragment: lane = column n (l%16), half g holds
// contiguous K = g*16 .. g*16+15.
DEV v16bf loadB(const __bf16* col, int hg) {
  const __bf16* p = col + hg * 16;
  return join16(*(const v8bf*)p, *(const v8bf*)(p + 8));
}

// Low 32 bits of a flat shared pointer == LDS byte offset (ISA 10.2 aperture).
DEV unsigned lds_off(const void* p) { return (unsigned)(size_t)p; }

// Async global->LDS copy of 16 bytes per lane (GVS mode: saddr + vaddr off).
DEV void async_copy_b128(const void* gbase, unsigned byte_off, unsigned lds_dst) {
  asm volatile("global_load_async_to_lds_b128 %0, %1, %2"
               :: "v"(lds_dst), "v"(byte_off),
                  "s"((unsigned long long)(size_t)gbase)
               : "memory");
}
DEV void wait_async0() { asm volatile("s_wait_asynccnt 0" ::: "memory"); }

// ---------------------------------------------------------------------------
// TDM: 2D tile load Global->LDS.  D# packing per ISA 8.3/8.4.
//   dim0 = contiguous elements (data_size = 2 bytes), dim1 = rows (stride0).
// 6-arg builtin: (u32x4 g0, i32x8 g1, i32x4 g2, i32x4 g3, i32x8 extra, cpol)
// ---------------------------------------------------------------------------
DEV void tdm_load_2d(unsigned lds_addr, const void* gptr,
                     unsigned tensor_d0, unsigned tensor_d1,
                     unsigned tile_d0, unsigned tile_d1,
                     unsigned long long stride0_elems) {
  unsigned long long ga = (unsigned long long)(size_t)gptr;
  u32x4 g0;
  g0[0] = 1u;                                           // count=1, user D#
  g0[1] = lds_addr;                                     // bits 63:32
  g0[2] = (unsigned)ga;                                 // global_addr[31:0]
  g0[3] = ((unsigned)(ga >> 32) & 0x1FFFFFFu) | (2u << 30);  // addr[56:32]|type=2
  i32x8 g1;
  g1[0] = (1 << 16);                                    // data_size=1 (2B), mask=0
  g1[1] = (int)((tensor_d0 & 0xFFFFu) << 16);           // tensor_dim0[15:0]
  g1[2] = (int)((tensor_d0 >> 16) | ((tensor_d1 & 0xFFFFu) << 16));
  g1[3] = (int)((tensor_d1 >> 16) | (tile_d0 << 16));   // tile_dim0
  g1[4] = (int)tile_d1;                                 // tile_dim1, tile_dim2=0
  g1[5] = (int)(unsigned)stride0_elems;                 // stride0[31:0]
  g1[6] = (int)((unsigned)(stride0_elems >> 32) & 0xFFFFu);  // stride0[47:32]
  g1[7] = 0;
  i32x4 z4 = {0, 0, 0, 0};
  i32x8 z8 = {0, 0, 0, 0, 0, 0, 0, 0};
  __builtin_amdgcn_tensor_load_to_lds(g0, g1, z4, z4, z8, 0);
}

// ---------------------------------------------------------------------------
// One-time f32 -> bf16 conversion (inputs and weights), 4 elems/thread.
// ---------------------------------------------------------------------------
__global__ __launch_bounds__(256)
void cvt_bf16(const float* __restrict__ src, unsigned short* __restrict__ dst,
              int n4) {
  int i = blockIdx.x * 256 + threadIdx.x;
  if (i >= n4) return;
  float4 f = ((const float4*)src)[i];
  uint2 u;
  u.x = pack2bf(f.x, f.y);
  u.y = pack2bf(f.z, f.w);
  ((uint2*)dst)[i] = u;
}

// ---------------------------------------------------------------------------
// Tiled GEMM  Y = X @ W^T (both bf16).  Block tile 128x128, 8 waves (2x4),
// BK=32.  Async global->LDS staging, double-buffered; the copy of K-slice i+1
// overlaps the WMMAs of slice i (one barrier per slice).
// MODE 0: bf16 out [B,H,S,dk] (Q/K proj; escale folds 1/sqrt(dk) into Q)
// MODE 1: bf16 out [B,H,dk,S] (V transposed; packed b128 stores along S)
// MODE 2: f32  out [4096,1024] (final projection)
// ---------------------------------------------------------------------------
template<int MODE>
__global__ __launch_bounds__(256)
void gemm128(const unsigned short* __restrict__ Xb,
             const unsigned short* __restrict__ Wb,
             void* __restrict__ Yv, float escale) {
  __shared__ unsigned short As[2][128 * 32];
  __shared__ unsigned short Bs[2][128 * 32];

  const int t    = threadIdx.x;
  const int lane = t & 31;
  const int w    = t >> 5;
  const int hg   = lane >> 4;
  const int ln   = lane & 15;
  const int wr   = w >> 2;
  const int wc   = w & 3;
  const int m0   = blockIdx.y * 128;
  const int n0   = blockIdx.x * 128;

  auto stage = [&](int k0, int buf) {
#pragma unroll
    for (int i = 0; i < 2; ++i) {
      int chunk = t + i * 256;              // 0..511 16B chunks
      int row   = chunk >> 2;
      int kk    = (chunk & 3) << 3;
      unsigned go_a = (unsigned)(((m0 + row) * Dc + k0 + kk) * 2);
      unsigned go_b = (unsigned)(((n0 + row) * Dc + k0 + kk) * 2);
      async_copy_b128(Xb, go_a, lds_off(&As[buf][row * 32 + kk]));
      async_copy_b128(Wb, go_b, lds_off(&Bs[buf][row * 32 + kk]));
    }
  };

  v8f acc[4][2];
#pragma unroll
  for (int rt = 0; rt < 4; ++rt)
#pragma unroll
    for (int ct = 0; ct < 2; ++ct)
      acc[rt][ct] = (v8f){0.f,0.f,0.f,0.f,0.f,0.f,0.f,0.f};

  stage(0, 0);
  int buf = 0;
  for (int k0 = 0; k0 < Dc; k0 += 32, buf ^= 1) {
    wait_async0();
    __syncthreads();
    if (k0 + 32 < Dc) stage(k0 + 32, buf ^ 1);

    v16bf bfrag[2];
#pragma unroll
    for (int ct = 0; ct < 2; ++ct) {
      int bn = wc * 32 + ct * 16 + ln;
      bfrag[ct] = loadB((const __bf16*)&Bs[buf][bn * 32], hg);
    }
#pragma unroll
    for (int rt = 0; rt < 4; ++rt) {
      int am = wr * 64 + rt * 16 + ln;
      v16bf af = loadA((const __bf16*)&As[buf][am * 32], hg);
#pragma unroll
      for (int ct = 0; ct < 2; ++ct)
        acc[rt][ct] = __builtin_amdgcn_wmma_f32_16x16x32_bf16(
            false, af, false, bfrag[ct], (short)0, acc[rt][ct], false, false);
    }
  }

#pragma unroll
  for (int rt = 0; rt < 4; ++rt)
#pragma unroll
    for (int ct = 0; ct < 2; ++ct) {
      int gmb = m0 + wr * 64 + rt * 16 + 8 * hg;        // first of 8 tokens
      int gn  = n0 + wc * 32 + ct * 16 + ln;            // output channel
      if (MODE == 1) {
        // tokens gmb..gmb+7 are consecutive; Vt index = ((b*H+h)*64+d)*S + s
        int b = gmb >> 11, s = gmb & (Sc - 1);
        int h = gn >> 6,  d = gn & (DKc - 1);
        uint4 st;
        st.x = pack2bf(acc[rt][ct][0] * escale, acc[rt][ct][1] * escale);
        st.y = pack2bf(acc[rt][ct][2] * escale, acc[rt][ct][3] * escale);
        st.z = pack2bf(acc[rt][ct][4] * escale, acc[rt][ct][5] * escale);
        st.w = pack2bf(acc[rt][ct][6] * escale, acc[rt][ct][7] * escale);
        *(uint4*)&((unsigned short*)Yv)[((size_t)(b * Hc + h) * DKc + d) * Sc + s] = st;
      } else {
#pragma unroll
        for (int r = 0; r < 8; ++r) {
          int gm = gmb + r;
          float v = acc[rt][ct][r];
          if (MODE == 2) {
            ((float*)Yv)[(size_t)gm * Dc + gn] = v;
          } else {
            int b = gm >> 11, s = gm & (Sc - 1);
            int h = gn >> 6,  d = gn & (DKc - 1);
            ((unsigned short*)Yv)[(((size_t)(b * Hc + h) * Sc + s) << 6) + d] =
                f2bf(v * escale);
          }
        }
      }
    }
}

// ---------------------------------------------------------------------------
// Flash-style attention (transposed).  8 waves/block; each wave owns 16 query
// columns of one (b,h).  Per 32-key chunk:
//   S^T(32x16) = K_tile * Q^T   (keys = rows, queries = lanes)
//   per-lane online softmax (1 shfl_xor(16) for max, 1 for sum)
//   P^T built in registers via half-wave exchange, O^T += V^T_tile * P^T
// K/V tiles are DMAd into LDS by the TDM, double-buffered.
// ---------------------------------------------------------------------------
__global__ __launch_bounds__(256)
void attn_kernel(const unsigned short* __restrict__ Qb,
                 const unsigned short* __restrict__ Kb,
                 const unsigned short* __restrict__ Vt,
                 const int* __restrict__ mask,
                 const int* __restrict__ pmask,   // [S,B,1]
                 unsigned short* __restrict__ ctx) {
  __shared__ unsigned short Kt[2][32 * DKc];   // [key][ch]   4KB each
  __shared__ unsigned short Vl[2][DKc * 32];   // [ch][key]   4KB each

  const int t    = threadIdx.x;
  const int lane = t & 31;
  const int w    = t >> 5;
  const int hg   = lane >> 4;
  const int ln   = lane & 15;
  const int bh   = blockIdx.x >> 4;
  const int qt   = blockIdx.x & 15;
  const int b    = bh >> 4;
  const int h    = bh & (Hc - 1);
  const int q    = qt * 128 + w * 16 + ln;     // this lane's query (fixed)

  const unsigned short* Qh = Qb + (size_t)bh * Sc * DKc;
  const unsigned short* Kh = Kb + (size_t)bh * Sc * DKc;
  const unsigned short* Vh = Vt + (size_t)bh * DKc * Sc;
  const int* mrow = mask + ((size_t)b * Sc + q) * Sc;

  auto stage_kv = [&](int kb, int buf) {
    // K tile: 32 rows of 64 contiguous ch elements, row stride 64
    tdm_load_2d(lds_off(&Kt[buf][0]), Kh + (size_t)kb * DKc,
                DKc, Sc, DKc, 32, DKc);
    // V tile: 64 rows of 32 contiguous key elements, row stride S
    tdm_load_2d(lds_off(&Vl[buf][0]), Vh + kb,
                Sc, DKc, 32, DKc, Sc);
  };

  // Q^T B-fragments (loop-invariant); Q was pre-scaled by 1/sqrt(dk).
  v16bf qfr[2];
  {
    const __bf16* qp = (const __bf16*)(Qh + (size_t)q * DKc);
#pragma unroll
    for (int c = 0; c < 2; ++c) qfr[c] = loadB(qp + c * 32, hg);
  }

  float mrun = -INFINITY, lrun = 0.f;
  v8f o[4];
#pragma unroll
  for (int n = 0; n < 4; ++n) o[n] = (v8f){0.f,0.f,0.f,0.f,0.f,0.f,0.f,0.f};

  if (w == 0) stage_kv(0, 0);
  int buf = 0;
  for (int kb = 0; kb < Sc; kb += 32, buf ^= 1) {
    if (w == 0) __builtin_amdgcn_s_wait_tensorcnt(0);
    __syncthreads();
    if (w == 0 && kb + 32 < Sc) stage_kv(kb + 32, buf ^ 1);

    // ---- S^T: two 16-key C frags (rows = keys, lane = query) --------------
    v8f s[2];
#pragma unroll
    for (int sub = 0; sub < 2; ++sub) {
      s[sub] = (v8f){0.f,0.f,0.f,0.f,0.f,0.f,0.f,0.f};
      const __bf16* kp = (const __bf16*)&Kt[buf][(sub * 16 + ln) * DKc];
#pragma unroll
      for (int c = 0; c < 2; ++c) {
        v16bf ka = loadA(kp + c * 32, hg);
        s[sub] = __builtin_amdgcn_wmma_f32_16x16x32_bf16(
            false, ka, false, qfr[c], (short)0, s[sub], false, false);
      }
    }
    // ---- mask: lane sees 8 consecutive keys per frag -> 2x b128 loads -----
    float v[2][8];
#pragma unroll
    for (int sub = 0; sub < 2; ++sub) {
      const int* mp = mrow + kb + sub * 16 + 8 * hg;
      int4 ma = ((const int4*)mp)[0];
      int4 mb = ((const int4*)mp)[1];
      int mk[8] = {ma.x, ma.y, ma.z, ma.w, mb.x, mb.y, mb.z, mb.w};
#pragma unroll
      for (int r = 0; r < 8; ++r)
        v[sub][r] = (mk[r] == 0) ? -1e9f : s[sub][r];
    }
    if (q == Sc - 1) {    // padding mask applies to last query row only
#pragma unroll
      for (int sub = 0; sub < 2; ++sub)
#pragma unroll
        for (int r = 0; r < 8; ++r) {
          int key = kb + sub * 16 + 8 * hg + r;
          if (pmask[(size_t)key * Bc + b] == 0) v[sub][r] = -1e9f;
        }
    }
    // ---- per-lane online softmax ------------------------------------------
    float mx = v[0][0];
#pragma unroll
    for (int sub = 0; sub < 2; ++sub)
#pragma unroll
      for (int r = 0; r < 8; ++r) mx = fmaxf(mx, v[sub][r]);
    mx = fmaxf(mx, __shfl_xor(mx, 16));        // combine the two half-waves
    float mn    = fmaxf(mrun, mx);
    float alpha = __expf(mrun - mn);
    mrun = mn;
    float p[2][8];
    float ps = 0.f;
#pragma unroll
    for (int sub = 0; sub < 2; ++sub)
#pragma unroll
      for (int r = 0; r < 8; ++r) {
        p[sub][r] = __expf(v[sub][r] - mn);
        ps += p[sub][r];
      }
    ps += __shfl_xor(ps, 16);
    lrun = lrun * alpha + ps;
#pragma unroll
    for (int n = 0; n < 4; ++n)
#pragma unroll
      for (int r = 0; r < 8; ++r) o[n][r] *= alpha;

    // ---- build P^T B-fragment in registers (half-wave exchange) -----------
    // B layout wants lane half g to hold keys g*16..g*16+15; this lane owns
    // keys {8*hg..8*hg+7} of each sub-tile; partner (lane^16) owns the rest.
    float lowv[8], highv[8];
#pragma unroll
    for (int r = 0; r < 8; ++r) {
      float send = hg ? p[0][r] : p[1][r];
      float recv = __shfl_xor(send, 16);
      lowv[r]  = hg ? recv : p[0][r];    // keys g*16 + 0..7
      highv[r] = hg ? p[1][r] : recv;    // keys g*16 + 8..15
    }
    union { unsigned u[8]; v16bf v; } pw;
#pragma unroll
    for (int j = 0; j < 4; ++j) {
      pw.u[j]     = pack2bf(lowv[2 * j],  lowv[2 * j + 1]);
      pw.u[4 + j] = pack2bf(highv[2 * j], highv[2 * j + 1]);
    }
    // ---- O^T += V^T_tile * P^T --------------------------------------------
#pragma unroll
    for (int n = 0; n < 4; ++n) {
      v16bf va = loadA((const __bf16*)&Vl[buf][(n * 16 + ln) * 32], hg);
      o[n] = __builtin_amdgcn_wmma_f32_16x16x32_bf16(
          false, va, false, pw.v, (short)0, o[n], false, false);
    }
  }

  // ---- normalize; each lane owns 8 contiguous channels per frag -----------
  float rinv = 1.f / lrun;
#pragma unroll
  for (int n = 0; n < 4; ++n) {
    uint4 st;
    st.x = pack2bf(o[n][0] * rinv, o[n][1] * rinv);
    st.y = pack2bf(o[n][2] * rinv, o[n][3] * rinv);
    st.z = pack2bf(o[n][4] * rinv, o[n][5] * rinv);
    st.w = pack2bf(o[n][6] * rinv, o[n][7] * rinv);
    int ch = n * 16 + 8 * hg;
    *(uint4*)&ctx[((size_t)(b * Sc + q)) * Dc + h * DKc + ch] = st;
  }
}

// ---------------------------------------------------------------------------
extern "C" void kernel_launch(void* const* d_in, const int* in_sizes, int n_in,
                              void* d_out, int out_size, void* d_ws, size_t ws_size,
                              hipStream_t stream) {
  const float* q    = (const float*)d_in[0];
  const float* k    = (const float*)d_in[1];
  const float* v    = (const float*)d_in[2];
  const int*   mask = (const int*)d_in[3];
  const int*   pm   = (const int*)d_in[4];
  const float* wq   = (const float*)d_in[5];
  const float* wk   = (const float*)d_in[6];
  const float* wv   = (const float*)d_in[7];
  const float* wo   = (const float*)d_in[8];
  float* out = (float*)d_out;

  unsigned short* ws = (unsigned short*)d_ws;
  const size_t tok = (size_t)Mtot * Dc;   // 4,194,304 elements
  const size_t wsz = (size_t)Dc * Dc;     // 1,048,576 elements
  unsigned short* Qb = ws;                // [B,H,S,dk]
  unsigned short* Kb = Qb + tok;
  unsigned short* Vt = Kb + tok;          // [B,H,dk,S]
  unsigned short* Cx = Vt + tok;          // ctx bf16 [4096,1024]
  unsigned short* Xq = Cx + tok;
  unsigned short* Xk = Xq + tok;
  unsigned short* Xv = Xk + tok;
  unsigned short* Wq = Xv + tok;
  unsigned short* Wk = Wq + wsz;
  unsigned short* Wv = Wk + wsz;
  unsigned short* Wo = Wv + wsz;

  const int tok4 = (int)(tok / 4), w4 = (int)(wsz / 4);
  dim3 cg((tok4 + 255) / 256);
  dim3 wg((w4 + 255) / 256);
  cvt_bf16<<<cg, 256, 0, stream>>>(q,  Xq, tok4);
  cvt_bf16<<<cg, 256, 0, stream>>>(k,  Xk, tok4);
  cvt_bf16<<<cg, 256, 0, stream>>>(v,  Xv, tok4);
  cvt_bf16<<<wg, 256, 0, stream>>>(wq, Wq, w4);
  cvt_bf16<<<wg, 256, 0, stream>>>(wk, Wk, w4);
  cvt_bf16<<<wg, 256, 0, stream>>>(wv, Wv, w4);
  cvt_bf16<<<wg, 256, 0, stream>>>(wo, Wo, w4);

  const float qscale = 0.125f;            // 1/sqrt(64) folded into Q
  dim3 grid(Dc / 128, Mtot / 128);        // (8, 32)
  gemm128<0><<<grid, 256, 0, stream>>>(Xq, Wq, Qb, qscale);
  gemm128<0><<<grid, 256, 0, stream>>>(Xk, Wk, Kb, 1.0f);
  gemm128<1><<<grid, 256, 0, stream>>>(Xv, Wv, Vt, 1.0f);
  attn_kernel<<<dim3(Bc * Hc * (Sc / 128)), 256, 0, stream>>>(
      Qb, Kb, Vt, mask, pm, Cx);
  gemm128<2><<<grid, 256, 0, stream>>>(Cx, Wo, out, 1.0f);
}
